// MultiHeadAttention_9509057593595
// MI455X (gfx1250) — matrix operand order repaired and checked by reading, not compiled, not run on hardware
//
#include <hip/hip_runtime.h>
#include <math.h>

// ---------------------------------------------------------------------------
// MI455X (gfx1250) fused multi-head attention, bf16 WMMA path.
//   stage 1: fp32 -> bf16 convert (Q,K,V, Wq,Wk,Wv,Wo)
//   stage 2: q/k/v = X @ W^T   (bf16 WMMA, f32 accum) -> [B,H,S,D] bf16
//   stage 3: flash attention (online softmax, scores NEVER materialized)
//            K tile staged via GLOBAL_LOAD_ASYNC_TO_LDS (ASYNCcnt)
//   stage 4: out = attn @ Wo^T + bo  (f32 output)
// NOTE: reference applies softmax to UNSCALED scores (bug reproduced): no 1/sqrt(D).
// ---------------------------------------------------------------------------

typedef __bf16 bf16;
typedef __bf16 v16bf __attribute__((ext_vector_type(16)));
typedef float  v8f   __attribute__((ext_vector_type(8)));
typedef int    v4i   __attribute__((ext_vector_type(4)));

#define B_   4
#define S_   2048
#define E_   1024
#define H_   16
#define D_   64
#define MTOT (B_ * S_)   // 8192

#if __has_builtin(__builtin_amdgcn_global_load_async_to_lds_b128)
#define HAVE_ASYNC_LDS 1
// clang prints AS(1) as "__device__" in HIP diagnostics; expected param type is
// "int __vector(4) __device__ *" == AS(1) v4i*. LDS side analogously AS(3) v4i*.
typedef __attribute__((address_space(1))) v4i* gas1_v4i_p;
typedef __attribute__((address_space(3))) v4i* las3_v4i_p;
#endif

// Load two 16-byte chunks (8 bf16 each) into one WMMA 16-element bf16 fragment.
__device__ __forceinline__ v16bf load2x16(const bf16* p0, const bf16* p1) {
  union { uint4 u[2]; v16bf v; } f;
  f.u[0] = *reinterpret_cast<const uint4*>(p0);
  f.u[1] = *reinterpret_cast<const uint4*>(p1);
  return f.v;
}

__device__ __forceinline__ v8f wmma_bf16(v16bf a, v16bf b, v8f c) {
  return __builtin_amdgcn_wmma_f32_16x16x32_bf16(false, a, false, b,
                                                 (short)0, c, false, false);
}

// ---------------------------------------------------------------------------
// Stage 1: fp32 -> bf16
// ---------------------------------------------------------------------------
__global__ void cvt_f32_bf16(const float* __restrict__ in,
                             bf16* __restrict__ out, int n) {
  int i = blockIdx.x * blockDim.x + threadIdx.x;
  int stride = gridDim.x * blockDim.x;
  for (; i < n; i += stride) out[i] = (bf16)in[i];
}

// ---------------------------------------------------------------------------
// Stages 2 & 4: C = A[M,1024] @ W[1024,1024]^T  (bf16 in, f32 accum)
//   mode 0: write bf16 head-split [B,H,S,D]  (QKV projections)
//   mode 1: write f32 [M,E] + bias           (output projection)
// Block: 256 threads = 8 waves; wave tile 32x32 (2x2 WMMA accs); WG tile 64x128.
// ---------------------------------------------------------------------------
__global__ void __launch_bounds__(256)
gemm_bf16_wmma(const bf16* __restrict__ A, const bf16* __restrict__ W,
               bf16* __restrict__ outHS, float* __restrict__ outF,
               const float* __restrict__ bias, int mode) {
  const int lane = threadIdx.x & 31;
  const int wave = threadIdx.x >> 5;
  const int hi   = lane >> 4;
  const int l16  = lane & 15;
  const int tileM = blockIdx.x * 64  + (wave & 1) * 32;
  const int tileN = blockIdx.y * 128 + (wave >> 1) * 32;

  v8f acc[2][2] = {};

  for (int k0 = 0; k0 < E_; k0 += 32) {
    v16bf af[2], wf[2];
#pragma unroll
    for (int mi = 0; mi < 2; ++mi) {
      const bf16* row = A + (size_t)(tileM + mi * 16 + l16) * E_ + k0 + hi * 8;
      af[mi] = load2x16(row, row + 16);
      __builtin_prefetch(row + 32, 0, 1);  // unconditional: speculative, branch-free
    }
#pragma unroll
    for (int ni = 0; ni < 2; ++ni) {
      const bf16* row = W + (size_t)(tileN + ni * 16 + l16) * E_ + k0 + hi * 8;
      wf[ni] = load2x16(row, row + 16);
    }
#pragma unroll
    for (int mi = 0; mi < 2; ++mi)
#pragma unroll
      for (int ni = 0; ni < 2; ++ni)
        acc[mi][ni] = wmma_bf16(af[mi], wf[ni], acc[mi][ni]);
  }

  // C layout: vgpr r -> M = r + 8*hi, N = lane%16
#pragma unroll
  for (int mi = 0; mi < 2; ++mi)
#pragma unroll
    for (int ni = 0; ni < 2; ++ni)
#pragma unroll
      for (int r = 0; r < 8; ++r) {
        const int m = tileM + mi * 16 + r + hi * 8;
        const int n = tileN + ni * 16 + l16;
        const float v = acc[mi][ni][r];
        if (mode == 0) {
          const int b = m >> 11, s = m & (S_ - 1);
          const int h = n >> 6,  d = n & (D_ - 1);
          outHS[((size_t)((b * H_ + h) * S_ + s)) * D_ + d] = (bf16)v;
        } else {
          outF[(size_t)m * E_ + n] = v + bias[n];
        }
      }
}

// ---------------------------------------------------------------------------
// Stage 3: flash attention.
// Grid: (S/128, B*H). 8 waves; wave owns 16 query rows. 32-key tiles from LDS.
// ---------------------------------------------------------------------------
__global__ void __launch_bounds__(256)
flash_attn_wmma(const bf16* __restrict__ qh, const bf16* __restrict__ kh,
                const bf16* __restrict__ vh, bf16* __restrict__ attn) {
  __shared__ bf16 ksh[32 * 64];        // [key][d]   row-major
  __shared__ bf16 vshT[64 * 32];       // [d][key]   transposed for PV B-frags
  __shared__ bf16 psh[8][16 * 32];     // per-wave P bounce (C-layout -> A-layout)

  const int lane = threadIdx.x & 31;
  const int wave = threadIdx.x >> 5;
  const int hi   = lane >> 4;
  const int l16  = lane & 15;
  const int bh   = blockIdx.y;              // b*H + h
  const int b    = bh >> 4, h = bh & 15;
  const size_t base = (size_t)bh * S_ * D_;
  const int q0 = blockIdx.x * 128 + wave * 16;

  // Q A-fragments for the two K-chunks d=[0,32) and d=[32,64)
  v16bf qf[2];
  {
    const bf16* qr = qh + base + (size_t)(q0 + l16) * D_ + hi * 8;
    qf[0] = load2x16(qr,      qr + 16);
    qf[1] = load2x16(qr + 32, qr + 48);
  }

  float mrow[8], lrow[8];
  v8f oacc[4] = {};
#pragma unroll
  for (int r = 0; r < 8; ++r) { mrow[r] = -1e30f; lrow[r] = 0.0f; }

  const int tid = threadIdx.x;
  const int ldr = tid >> 3;          // key row 0..31 in tile
  const int ldc = (tid & 7) * 8;     // d chunk

  for (int kt = 0; kt < S_ / 32; ++kt) {
    // ---- cooperative stage-in: K row-major (async->LDS), V transposed ----
    const size_t gsrc = base + (size_t)(kt * 32 + ldr) * D_ + ldc;
#ifdef HAVE_ASYNC_LDS
    // Integer-routed casts: AS(1) ptr is 64-bit (bit-identical); AS(3) ptr is
    // 32-bit and the low 32 bits of a generic LDS address ARE the DS offset
    __builtin_amdgcn_global_load_async_to_lds_b128(
        (gas1_v4i_p)(unsigned long long)(const void*)&kh[gsrc],
        (las3_v4i_p)(unsigned int)(unsigned long long)(void*)&ksh[ldr * 64 + ldc],
        0, 0);
#else
    *reinterpret_cast<uint4*>(&ksh[ldr * 64 + ldc]) =
        *reinterpret_cast<const uint4*>(&kh[gsrc]);
#endif
    {
      union { uint4 u; bf16 e[8]; } vv;
      vv.u = *reinterpret_cast<const uint4*>(&vh[gsrc]);
#pragma unroll
      for (int j = 0; j < 8; ++j) vshT[(ldc + j) * 32 + ldr] = vv.e[j];
    }
#ifdef HAVE_ASYNC_LDS
#if __has_builtin(__builtin_amdgcn_s_wait_asynccnt)
    __builtin_amdgcn_s_wait_asynccnt(0);
#else
    asm volatile("s_wait_asynccnt 0x0" ::: "memory");
#endif
#endif
    __syncthreads();

    // ---- S = Q @ K^T : two 16x16 key subtiles, 2 chained WMMAs each ----
    v8f sac[2] = {};
#pragma unroll
    for (int ni = 0; ni < 2; ++ni) {
      const bf16* kr = &ksh[(ni * 16 + l16) * 64];
      v16bf kf0 = load2x16(kr + hi * 8,      kr + hi * 8 + 16);       // d 0..31
      v16bf kf1 = load2x16(kr + 32 + hi * 8, kr + 32 + hi * 8 + 16);  // d 32..63
      sac[ni] = wmma_bf16(qf[0], kf0, sac[ni]);
      sac[ni] = wmma_bf16(qf[1], kf1, sac[ni]);
    }

    // ---- online softmax (row stats live per (vgpr r, lane-half)) ----
    bf16* pw = psh[wave];
#pragma unroll
    for (int r = 0; r < 8; ++r) {
      float rmax = fmaxf(sac[0][r], sac[1][r]);
#pragma unroll
      for (int mk = 1; mk < 16; mk <<= 1)
        rmax = fmaxf(rmax, __shfl_xor(rmax, mk, 32));
      const float mnew = fmaxf(mrow[r], rmax);
      const float scl  = __expf(mrow[r] - mnew);
      const float e0   = __expf(sac[0][r] - mnew);
      const float e1   = __expf(sac[1][r] - mnew);
      float rs = e0 + e1;
#pragma unroll
      for (int mk = 1; mk < 16; mk <<= 1) rs += __shfl_xor(rs, mk, 32);
      lrow[r] = lrow[r] * scl + rs;
      mrow[r] = mnew;
#pragma unroll
      for (int ds = 0; ds < 4; ++ds) oacc[ds][r] *= scl;
      const int m = r + hi * 8;
      pw[m * 32 + l16]      = (bf16)e0;   // C-layout -> row-major P in LDS
      pw[m * 32 + 16 + l16] = (bf16)e1;
    }

    // ---- O += P @ V : P re-read in A-layout, V^T gives contiguous B-frags ----
    const bf16* prow = &pw[l16 * 32];
    v16bf pf = load2x16(prow + hi * 8, prow + 16 + hi * 8);
#pragma unroll
    for (int ds = 0; ds < 4; ++ds) {
      const bf16* vr = &vshT[(ds * 16 + l16) * 32 + hi * 8];
      v16bf vf = load2x16(vr, vr + 16);
      oacc[ds] = wmma_bf16(pf, vf, oacc[ds]);
    }
    __syncthreads();
  }

  // ---- epilogue: normalize, merge heads into [B,S,E] bf16 ----
#pragma unroll
  for (int ds = 0; ds < 4; ++ds)
#pragma unroll
    for (int r = 0; r < 8; ++r) {
      const int srow = q0 + r + hi * 8;
      const int d    = ds * 16 + l16;
      const float v  = oacc[ds][r] / lrow[r];
      attn[((size_t)(b * S_ + srow)) * E_ + h * D_ + d] = (bf16)v;
    }
}

// ---------------------------------------------------------------------------
extern "C" void kernel_launch(void* const* d_in, const int* in_sizes, int n_in,
                              void* d_out, int out_size, void* d_ws, size_t ws_size,
                              hipStream_t stream) {
  (void)in_sizes; (void)n_in; (void)out_size; (void)ws_size;
  const float* Q  = (const float*)d_in[0];
  const float* K  = (const float*)d_in[1];
  const float* V  = (const float*)d_in[2];
  const float* Wq = (const float*)d_in[3];
  const float* Wk = (const float*)d_in[4];
  const float* Wv = (const float*)d_in[5];
  const float* Wo = (const float*)d_in[6];
  const float* bo = (const float*)d_in[7];
  float* out = (float*)d_out;

  const size_t nQ = (size_t)B_ * S_ * E_;  // 8Mi
  const size_t nW = (size_t)E_ * E_;       // 1Mi
  bf16* ws   = (bf16*)d_ws;
  bf16* Qb   = ws;
  bf16* Kb   = Qb + nQ;
  bf16* Vb   = Kb + nQ;
  bf16* Wqb  = Vb + nQ;
  bf16* Wkb  = Wqb + nW;
  bf16* Wvb  = Wkb + nW;
  bf16* Wob  = Wvb + nW;
  bf16* qhd  = Wob + nW;   // [B,H,S,D]
  bf16* khd  = qhd + nQ;
  bf16* vhd  = khd + nQ;
  bf16* attn = vhd + nQ;   // [B,S,E]

  // Stage 1: convert
  cvt_f32_bf16<<<2048, 256, 0, stream>>>(Q,  Qb,  (int)nQ);
  cvt_f32_bf16<<<2048, 256, 0, stream>>>(K,  Kb,  (int)nQ);
  cvt_f32_bf16<<<2048, 256, 0, stream>>>(V,  Vb,  (int)nQ);
  cvt_f32_bf16<<<512,  256, 0, stream>>>(Wq, Wqb, (int)nW);
  cvt_f32_bf16<<<512,  256, 0, stream>>>(Wk, Wkb, (int)nW);
  cvt_f32_bf16<<<512,  256, 0, stream>>>(Wv, Wvb, (int)nW);
  cvt_f32_bf16<<<512,  256, 0, stream>>>(Wo, Wob, (int)nW);

  // Stage 2: QKV projections (head-split bf16 out)
  dim3 ggrid(MTOT / 64, E_ / 128);
  gemm_bf16_wmma<<<ggrid, 256, 0, stream>>>(Qb, Wqb, qhd, nullptr, nullptr, 0);
  gemm_bf16_wmma<<<ggrid, 256, 0, stream>>>(Kb, Wkb, khd, nullptr, nullptr, 0);
  gemm_bf16_wmma<<<ggrid, 256, 0, stream>>>(Vb, Wvb, vhd, nullptr, nullptr, 0);

  // Stage 3: flash attention
  flash_attn_wmma<<<dim3(S_ / 128, B_ * H_), 256, 0, stream>>>(qhd, khd, vhd, attn);

  // Stage 4: output projection + bias (fp32 out)
  gemm_bf16_wmma<<<ggrid, 256, 0, stream>>>(attn, Wob, nullptr, out, bo, 1);
}
